// SequentialTransductionUnit_48395691491336
// MI455X (gfx1250) — compile-verified, matrix operand order checked by reading, not run on hardware
//
#include <hip/hip_runtime.h>

// ---------------------------------------------------------------------------
// HSTU SequentialTransductionUnit for MI455X (gfx1250), wave32 + WMMA f16.
// v3: 32x64 wave tiles (2x WMMA per B fragment), async-to-LDS A staging
//     (GLOBAL_LOAD_ASYNC_TO_LDS_B128 + s_wait_asynccnt), pipelined K loops,
//     rcp-based silu epilogues.
// ---------------------------------------------------------------------------

typedef __attribute__((ext_vector_type(16))) _Float16 v16h;
typedef __attribute__((ext_vector_type(8)))  _Float16 v8h;
typedef __attribute__((ext_vector_type(8)))  float    v8f;
typedef __attribute__((ext_vector_type(4)))  float    v4f;

#define B_   4
#define L_   1024
#define D_   512
#define H_   8
#define HD_  64
#define NB_  128
#define BL_  (B_ * L_)

// silu via v_rcp_f32 (avoids v_div_scale IEEE-division chains)
__device__ __forceinline__ float siluf(float x) {
  return x * __builtin_amdgcn_rcpf(1.0f + __expf(-x));
}

// ---------------------------------------------------------------------------
// WMMA 16x16x32 f16 fragment loaders (ISA 7.12.2 layouts).
// A (16x32, MxK) per lane m<16: halves 0..7 = K 0..7, halves 8..15 = K 16..23;
// lane m+16: K 8..15 and 24..31.  B (32x16, KxN) is the same pattern applied
// to column n == row n of the transposed operand, so one loader serves both.
// ---------------------------------------------------------------------------
__device__ __forceinline__ v16h load_frag_h(const _Float16* __restrict__ base,
                                            int ld, int row0, int k0, int lane) {
  int m  = row0 + (lane & 15);
  int kh = (lane >> 4) << 3;                 // 0 or 8
  const _Float16* p = base + (size_t)m * ld + k0 + kh;
  v8h a0 = *(const v8h*)p;                   // K chunk {k0+kh .. +7}
  v8h a1 = *(const v8h*)(p + 16);            // K chunk {k0+16+kh .. +7}
  v16h r;
#pragma unroll
  for (int e = 0; e < 8; ++e) { r[e] = a0[e]; r[e + 8] = a1[e]; }
  return r;
}

// Same fragment but sourced from fp32 memory (converted to f16 in registers).
__device__ __forceinline__ v16h load_frag_f(const float* __restrict__ base,
                                            int ld, int row0, int k0, int lane) {
  int m  = row0 + (lane & 15);
  int kh = (lane >> 4) << 3;
  const float* p = base + (size_t)m * ld + k0 + kh;
  v4f a0 = *(const v4f*)p;        v4f a1 = *(const v4f*)(p + 4);
  v4f b0 = *(const v4f*)(p + 16); v4f b1 = *(const v4f*)(p + 20);
  v16h r;
#pragma unroll
  for (int e = 0; e < 4; ++e) {
    r[e]      = (_Float16)a0[e];  r[e + 4]  = (_Float16)a1[e];
    r[e + 8]  = (_Float16)b0[e];  r[e + 12] = (_Float16)b1[e];
  }
  return r;
}

// ---------------------------------------------------------------------------
// 1) RMSNorm(hidden) * ln_w -> f16.  One 256-thread block per row of 512.
// ---------------------------------------------------------------------------
__global__ void rmsnorm_f16_kernel(const float* __restrict__ x,
                                   const float* __restrict__ w,
                                   _Float16* __restrict__ out) {
  int row = blockIdx.x;
  const float* xr = x + (size_t)row * D_;
  __shared__ float red[256];
  float a = xr[threadIdx.x];
  float b = xr[threadIdx.x + 256];
  red[threadIdx.x] = a * a + b * b;
  __syncthreads();
  for (int s = 128; s > 0; s >>= 1) {
    if (threadIdx.x < s) red[threadIdx.x] += red[threadIdx.x + s];
    __syncthreads();
  }
  float scale = rsqrtf(red[0] * (1.0f / D_) + 1e-6f);
  out[(size_t)row * D_ + threadIdx.x]       = (_Float16)(a * scale * w[threadIdx.x]);
  out[(size_t)row * D_ + threadIdx.x + 256] = (_Float16)(b * scale * w[threadIdx.x + 256]);
}

// ---------------------------------------------------------------------------
// f32 -> f16 elementwise (weight conversion).
// ---------------------------------------------------------------------------
__global__ void cvt_f16_kernel(const float* __restrict__ x, _Float16* __restrict__ y, int n) {
  int i = blockIdx.x * 256 + threadIdx.x;
  if (i < n) y[i] = (_Float16)x[i];
}

// ---------------------------------------------------------------------------
// Shared GEMM core: block = 8 waves; wave w owns the 32x64 tile
// (rows blockIdx.x*32, cols w*64).  The 32x512 f16 A tile (32 KB; WGP LDS is
// 320 KB) is staged via GLOBAL_LOAD_ASYNC_TO_LDS_B128 (ASYNCcnt path, no VGPR
// round-trip): each thread issues 8 async B128s covering its 128 B, then
// s_wait_asynccnt 0 + barrier.  A fragments come from ds_load_b128; B
// fragments are software-pipelined from global; each B fragment feeds TWO
// WMMAs (rows 0..15 and 16..31), doubling math per byte loaded.
// ---------------------------------------------------------------------------
__device__ __forceinline__ void gemm_core(const _Float16* __restrict__ A,
                                          const _Float16* __restrict__ W,
                                          _Float16* __restrict__ As,   // LDS 32*512
                                          int row0, v8f acc[2][4]) {
  // async stage of A tile: 256 threads x 128B (8 x B128 each)
  {
    int t = threadIdx.x;
    int r = t >> 3;                 // 0..31
    int c = (t & 7) * 64;           // halves; 64 halves = 128B per thread
    unsigned lds_addr = (unsigned)(size_t)(As + r * D_ + c);          // LDS byte addr (low 32b)
    unsigned long long gaddr = (unsigned long long)(size_t)(A + (size_t)(row0 + r) * D_ + c);
    asm volatile(
        "global_load_async_to_lds_b128 %0, %1, off\n\t"
        "global_load_async_to_lds_b128 %0, %1, off offset:16\n\t"
        "global_load_async_to_lds_b128 %0, %1, off offset:32\n\t"
        "global_load_async_to_lds_b128 %0, %1, off offset:48\n\t"
        "global_load_async_to_lds_b128 %0, %1, off offset:64\n\t"
        "global_load_async_to_lds_b128 %0, %1, off offset:80\n\t"
        "global_load_async_to_lds_b128 %0, %1, off offset:96\n\t"
        "global_load_async_to_lds_b128 %0, %1, off offset:112\n\t"
        "s_wait_asynccnt 0x0"
        :: "v"(lds_addr), "v"(gaddr) : "memory");
  }
  __syncthreads();

  int lane = threadIdx.x & 31;
  int wave = threadIdx.x >> 5;
  int col0 = wave * 64;
  int m  = lane & 15;
  int kh = (lane >> 4) << 3;

  // A fragment from LDS (rbase = 0 or 16 selects the row half)
  auto lds_frag = [&](int rbase, int k0) {
    v8h a0 = *(const v8h*)&As[(rbase + m) * D_ + k0 + kh];
    v8h a1 = *(const v8h*)&As[(rbase + m) * D_ + k0 + 16 + kh];
    v16h r;
#pragma unroll
    for (int e = 0; e < 8; ++e) { r[e] = a0[e]; r[e + 8] = a1[e]; }
    return r;
  };

  v16h a_cur[2];
  a_cur[0] = lds_frag(0, 0);
  a_cur[1] = lds_frag(16, 0);
  v16h b_cur[4];
#pragma unroll
  for (int t = 0; t < 4; ++t) b_cur[t] = load_frag_h(W, D_, col0 + 16 * t, 0, lane);

  for (int k0 = 0; k0 < D_; k0 += 32) {
    int kn = (k0 + 32 < D_) ? k0 + 32 : 0;   // last iter reloads k0=0 (discarded)
    v16h a_nxt[2];
    a_nxt[0] = lds_frag(0, kn);
    a_nxt[1] = lds_frag(16, kn);
    v16h b_nxt[4];
#pragma unroll
    for (int t = 0; t < 4; ++t) b_nxt[t] = load_frag_h(W, D_, col0 + 16 * t, kn, lane);
#pragma unroll
    for (int t = 0; t < 4; ++t) {
      acc[0][t] = __builtin_amdgcn_wmma_f32_16x16x32_f16(false, a_cur[0], false, b_cur[t],
                                                         (short)0, acc[0][t], false, false);
      acc[1][t] = __builtin_amdgcn_wmma_f32_16x16x32_f16(false, a_cur[1], false, b_cur[t],
                                                         (short)0, acc[1][t], false, false);
    }
    a_cur[0] = a_nxt[0];
    a_cur[1] = a_nxt[1];
#pragma unroll
    for (int t = 0; t < 4; ++t) b_cur[t] = b_nxt[t];
  }
}

// 2) Y = silu(A @ W^T) in f16
__global__ void gemm_silu_kernel(const _Float16* __restrict__ A,
                                 const _Float16* __restrict__ W,
                                 _Float16* __restrict__ Y) {
  __shared__ _Float16 As[32 * D_];
  int row0 = blockIdx.x * 32;
  v8f acc[2][4] = {};
  gemm_core(A, W, As, row0, acc);
  int lane = threadIdx.x & 31;
  int col0 = (threadIdx.x >> 5) * 64;
  int c = lane & 15;
  int r = (lane >> 4) << 3;
#pragma unroll
  for (int hf = 0; hf < 2; ++hf)
#pragma unroll
    for (int t = 0; t < 4; ++t)
#pragma unroll
      for (int g = 0; g < 8; ++g)
        Y[(size_t)(row0 + 16 * hf + r + g) * D_ + col0 + 16 * t + c] =
            (_Float16)siluf(acc[hf][t][g]);
}

// 7) out = residual + gated @ wo^T   (fp32 out)
__global__ void gemm_out_kernel(const _Float16* __restrict__ G,
                                const _Float16* __restrict__ Wo,
                                const float* __restrict__ resid,
                                float* __restrict__ out) {
  __shared__ _Float16 As[32 * D_];
  int row0 = blockIdx.x * 32;
  v8f acc[2][4] = {};
  gemm_core(G, Wo, As, row0, acc);
  int lane = threadIdx.x & 31;
  int col0 = (threadIdx.x >> 5) * 64;
  int c = lane & 15;
  int r = (lane >> 4) << 3;
#pragma unroll
  for (int hf = 0; hf < 2; ++hf)
#pragma unroll
    for (int t = 0; t < 4; ++t)
#pragma unroll
      for (int g = 0; g < 8; ++g) {
        size_t idx = (size_t)(row0 + 16 * hf + r + g) * D_ + col0 + 16 * t + c;
        out[idx] = resid[idx] + acc[hf][t][g];
      }
}

// ---------------------------------------------------------------------------
// 3) vt[b,h,d,j] = v[b,j,h,d]  (per-head transpose so attn@v B-frags are
// contiguous along K=j).
// ---------------------------------------------------------------------------
__global__ void transpose_v_kernel(const _Float16* __restrict__ v, _Float16* __restrict__ vt) {
  int idx = blockIdx.x * 256 + threadIdx.x;     // flat over B*L*D = 2^21
  int d = idx & (HD_ - 1);
  int h = (idx >> 6) & (H_ - 1);
  int j = (idx >> 9) & (L_ - 1);
  int b = idx >> 19;
  vt[(((size_t)(b * H_ + h)) * HD_ + d) * L_ + j] = v[idx];
}

// ---------------------------------------------------------------------------
// 4) attn[b,h,i,j] = causal * silu(q.k + ts_bias + pos_bias) / L  (fp32 out).
// One wave per 16x16 (i,j) tile; both K-steps' loads issued before the WMMAs.
// ---------------------------------------------------------------------------
__global__ void attn_qk_kernel(const _Float16* __restrict__ q16,
                               const _Float16* __restrict__ k16,
                               const int*      __restrict__ ts,
                               const float*    __restrict__ ts_w,
                               const float*    __restrict__ pos_w,
                               float*          __restrict__ attn) {
  int lane = threadIdx.x & 31;
  int wave = threadIdx.x >> 5;
  int jt = blockIdx.x * 8 + wave;
  int it = blockIdx.y;
  int bh = blockIdx.z;
  int b = bh >> 3, h = bh & 7;
  int i0 = it * 16, j0 = jt * 16;
  float* out = attn + ((size_t)bh * L_ + i0) * L_;
  int c = lane & 15;
  int r = (lane >> 4) << 3;
  if (j0 > i0 + 15) {                           // entirely above diagonal
#pragma unroll
    for (int g = 0; g < 8; ++g) out[(size_t)(r + g) * L_ + j0 + c] = 0.0f;
    return;
  }
  const _Float16* qb = q16 + ((size_t)b * L_) * D_ + h * HD_;
  const _Float16* kb = k16 + ((size_t)b * L_) * D_ + h * HD_;
  // issue all four fragment loads before the WMMAs
  v16h a0 = load_frag_h(qb, D_, i0, 0,  lane);
  v16h b0 = load_frag_h(kb, D_, j0, 0,  lane);
  v16h a1 = load_frag_h(qb, D_, i0, 32, lane);
  v16h b1 = load_frag_h(kb, D_, j0, 32, lane);
  v8f acc = {};
  acc = __builtin_amdgcn_wmma_f32_16x16x32_f16(false, a0, false, b0, (short)0, acc, false, false);
  acc = __builtin_amdgcn_wmma_f32_16x16x32_f16(false, a1, false, b1, (short)0, acc, false, false);

  const int* tsb = ts + b * L_;
  int j  = j0 + c;
  int tj = tsb[j];
#pragma unroll
  for (int g = 0; g < 8; ++g) {
    int i   = i0 + r + g;
    int ip1 = (i + 1 < L_) ? i + 1 : L_ - 1;    // ext[b,i+1]
    float ad = fabsf((float)(tsb[ip1] - tj));
    ad = ad < 1.0f ? 1.0f : ad;
    int bucket = (int)(__logf(ad) * (1.0f / 0.301f));
    bucket = bucket < 0 ? 0 : (bucket > NB_ ? NB_ : bucket);
    float bias = ts_w[bucket] + pos_w[(L_ - 1) + j - i];
    float val  = siluf(acc[g] + bias) * (1.0f / (float)L_);
    out[(size_t)(r + g) * L_ + j] = (j <= i) ? val : 0.0f;
  }
}

// ---------------------------------------------------------------------------
// 5) attn_out[b,i,h,d] = sum_j attn[b,h,i,j] * v[b,j,h,d].
// A-frags read attn fp32 (from d_out) and convert; B-frags from vt.
// Causal bound on K; next-iteration fragments prefetched before the WMMA.
// ---------------------------------------------------------------------------
__global__ void attn_v_kernel(const float* __restrict__ attn,
                              const _Float16* __restrict__ vt,
                              float* __restrict__ ao) {
  int lane = threadIdx.x & 31;
  int gw = blockIdx.x * 8 + (threadIdx.x >> 5); // [0, 8192)
  int dt = gw & 3;
  int it = (gw >> 2) & 63;
  int h  = (gw >> 8) & 7;
  int b  = gw >> 11;
  int bh = b * H_ + h;
  int i0 = it * 16, d0 = dt * 16;
  const float*    ab = attn + (size_t)bh * L_ * L_;
  const _Float16* vb = vt + (size_t)bh * HD_ * L_;
  int kmax = ((i0 + 16 + 31) >> 5) << 5;        // cover the diagonal tile
  v8f acc = {};
  v16h a_cur = load_frag_f(ab, L_, i0, 0, lane);
  v16h b_cur = load_frag_h(vb, L_, d0, 0, lane);
  for (int k0 = 0; k0 < kmax; k0 += 32) {
    int kn = (k0 + 32 < kmax) ? k0 + 32 : 0;    // last iter reloads 0 (discarded)
    v16h a_nxt = load_frag_f(ab, L_, i0, kn, lane);
    v16h b_nxt = load_frag_h(vb, L_, d0, kn, lane);
    acc = __builtin_amdgcn_wmma_f32_16x16x32_f16(false, a_cur, false, b_cur,
                                                 (short)0, acc, false, false);
    a_cur = a_nxt;
    b_cur = b_nxt;
  }
  int c = lane & 15;
  int r = (lane >> 4) << 3;
#pragma unroll
  for (int g = 0; g < 8; ++g)
    ao[((size_t)b * L_ + i0 + r + g) * D_ + h * HD_ + d0 + c] = acc[g];
}

// ---------------------------------------------------------------------------
// 6) gated = u * rmsnorm(attn_out, attn_ln_w)  -> f16
// ---------------------------------------------------------------------------
__global__ void gate_kernel(const float* __restrict__ ao,
                            const _Float16* __restrict__ u16,
                            const float* __restrict__ w,
                            _Float16* __restrict__ gated) {
  int row = blockIdx.x;
  const float* ar = ao + (size_t)row * D_;
  __shared__ float red[256];
  float a = ar[threadIdx.x];
  float b = ar[threadIdx.x + 256];
  red[threadIdx.x] = a * a + b * b;
  __syncthreads();
  for (int s = 128; s > 0; s >>= 1) {
    if (threadIdx.x < s) red[threadIdx.x] += red[threadIdx.x + s];
    __syncthreads();
  }
  float scale = rsqrtf(red[0] * (1.0f / D_) + 1e-6f);
  size_t o = (size_t)row * D_;
  gated[o + threadIdx.x] =
      (_Float16)((float)u16[o + threadIdx.x] * a * scale * w[threadIdx.x]);
  gated[o + threadIdx.x + 256] =
      (_Float16)((float)u16[o + threadIdx.x + 256] * b * scale * w[threadIdx.x + 256]);
}

// ---------------------------------------------------------------------------
// Launch
// ---------------------------------------------------------------------------
extern "C" void kernel_launch(void* const* d_in, const int* in_sizes, int n_in,
                              void* d_out, int out_size, void* d_ws, size_t ws_size,
                              hipStream_t stream) {
  (void)in_sizes; (void)n_in; (void)out_size; (void)ws_size;
  const float* hs    = (const float*)d_in[0];
  // d_in[1] = attention_mask (causal; computed analytically, unused)
  const int*   ts    = (const int*)d_in[2];
  const float* ln_w  = (const float*)d_in[3];
  const float* aln_w = (const float*)d_in[4];
  const float* wq    = (const float*)d_in[5];
  const float* wk    = (const float*)d_in[6];
  const float* wv    = (const float*)d_in[7];
  const float* wu    = (const float*)d_in[8];
  const float* wo    = (const float*)d_in[9];
  const float* ts_w  = (const float*)d_in[10];
  const float* pos_w = (const float*)d_in[11];

  // Workspace layout (bytes)
  char* ws = (char*)d_ws;
  _Float16* h16   = (_Float16*)(ws + (size_t)0);
  _Float16* q16   = (_Float16*)(ws + ((size_t)4  << 20));
  _Float16* k16   = (_Float16*)(ws + ((size_t)8  << 20));
  _Float16* v16   = (_Float16*)(ws + ((size_t)12 << 20));
  _Float16* u16   = (_Float16*)(ws + ((size_t)16 << 20));
  _Float16* vt16  = (_Float16*)(ws + ((size_t)20 << 20));
  float*    ao    = (float*)   (ws + ((size_t)24 << 20));
  _Float16* g16   = (_Float16*)(ws + ((size_t)32 << 20));
  _Float16* wq16  = (_Float16*)(ws + ((size_t)36 << 20));
  _Float16* wk16  = wq16 + (size_t)D_ * D_;
  _Float16* wv16  = wk16 + (size_t)D_ * D_;
  _Float16* wu16  = wv16 + (size_t)D_ * D_;
  _Float16* wo16  = wu16 + (size_t)D_ * D_;

  float* out_f  = (float*)d_out;
  float* attn_f = out_f + (size_t)BL_ * D_;    // [B,H,L,L]

  const int WN = D_ * D_;                      // 262144 elems per weight

  // 1) h = rmsnorm(x) * ln_w  -> f16
  rmsnorm_f16_kernel<<<BL_, 256, 0, stream>>>(hs, ln_w, h16);

  // weight conversions
  cvt_f16_kernel<<<(WN + 255) / 256, 256, 0, stream>>>(wq, wq16, WN);
  cvt_f16_kernel<<<(WN + 255) / 256, 256, 0, stream>>>(wk, wk16, WN);
  cvt_f16_kernel<<<(WN + 255) / 256, 256, 0, stream>>>(wv, wv16, WN);
  cvt_f16_kernel<<<(WN + 255) / 256, 256, 0, stream>>>(wu, wu16, WN);
  cvt_f16_kernel<<<(WN + 255) / 256, 256, 0, stream>>>(wo, wo16, WN);

  // 2) q/k/v/u = silu(h @ w^T)
  gemm_silu_kernel<<<BL_ / 32, 256, 0, stream>>>(h16, wq16, q16);
  gemm_silu_kernel<<<BL_ / 32, 256, 0, stream>>>(h16, wk16, k16);
  gemm_silu_kernel<<<BL_ / 32, 256, 0, stream>>>(h16, wv16, v16);
  gemm_silu_kernel<<<BL_ / 32, 256, 0, stream>>>(h16, wu16, u16);

  // 3) per-head transpose of v
  transpose_v_kernel<<<(BL_ * D_) / 256, 256, 0, stream>>>(v16, vt16);

  // 4) attn scores (writes the attn output region of d_out)
  attn_qk_kernel<<<dim3(8, L_ / 16, B_ * H_), 256, 0, stream>>>(
      q16, k16, ts, ts_w, pos_w, attn_f);

  // 5) attn @ v
  attn_v_kernel<<<(B_ * H_ * (L_ / 16) * (HD_ / 16)) / 8, 256, 0, stream>>>(
      attn_f, vt16, ao);

  // 6) gated = u * rmsnorm(attn_out)
  gate_kernel<<<BL_, 256, 0, stream>>>(ao, u16, aln_w, g16);

  // 7) out = residual + gated @ wo^T
  gemm_out_kernel<<<BL_ / 32, 256, 0, stream>>>(g16, wo16, hs, out_f);
}
